// PAM_Module_90211493085853
// MI455X (gfx1250) — matrix operand order repaired and checked by reading, not compiled, not run on hardware
//
#include <hip/hip_runtime.h>

#define CH   1024
#define HW   49
#define HWP  64    // padded hw (49 -> 64)
#define CK   32    // channels per LDS chunk
#define NTHREADS 256
#define FRAG 512   // bf16 elements per fragment set (32 lanes x 16)

typedef __attribute__((ext_vector_type(16))) __bf16 v16bf;
typedef __attribute__((ext_vector_type(8)))  float  v8f;

__device__ __forceinline__ __bf16 f2bf(float f) {
  union { float f; unsigned u; } x; x.f = f;
  unsigned r = (x.u + 0x7FFFu + ((x.u >> 16) & 1u)) >> 16;
  unsigned short h = (unsigned short)r;
  __bf16 b;
  __builtin_memcpy(&b, &h, 2);
  return b;
}

// Slot (lane*16 + elem) of element (m, k) inside a 16x32 bf16 A fragment.
// Contiguous k order used here: lane L -> M = L%16 ; elem j holds K = (L/16)*16 + j.
__device__ __forceinline__ int a_slot(int m, int k) {
  return ((((k >> 4) << 4) | m) << 4) | (k & 15);
}

// Slot of element (k, n) inside a 32x16 bf16 B fragment.
// Contiguous k order used here: lane L -> N = L%16 ; elem j holds K = (L/16)*16 + j.
__device__ __forceinline__ int b_slot(int k, int n) {
  return ((((k >> 4) << 4) | n) << 4) | (k & 15);
}

__global__ __launch_bounds__(NTHREADS)
void pam_kernel(const float* __restrict__ v1, const float* __restrict__ q1,
                const float* __restrict__ k1, float* __restrict__ out) {
  // Fragment-layout staging: one 32B contiguous fragment per lane -> 2x ds_load_b128.
  __shared__ __align__(32) __bf16 frag_q  [4 * FRAG];  // A frags (Q^T), per i-tile
  __shared__ __align__(32) __bf16 frag_k  [4 * FRAG];  // B frags (K),   per j-tile
  __shared__ __align__(32) __bf16 frag_v  [4 * FRAG];  // A frags (V),   per (ct,kk)
  __shared__ __align__(32) __bf16 frag_att[8 * FRAG];  // B frags (att), per (kk,jt)
  __shared__ float lds_att[HWP * HWP];                 // f32 att (softmax workspace)
  __shared__ float lds_vf [CK * HWP];                  // f32 V chunk (async-copy target)
  __shared__ float red[8];

  const int tid  = threadIdx.x;
  const int wv   = tid >> 5;
  const int lane = tid & 31;
  const int b    = blockIdx.x;
  const size_t base = (size_t)b * CH * HW;

  // Zero all fragment buffers once: pad rows/cols (i,j >= 49) stay exactly zero.
  for (int i = tid; i < 20 * FRAG; i += NTHREADS) {
    if (i < 4 * FRAG)            frag_q[i] = f2bf(0.f);
    else if (i < 8 * FRAG)       frag_k[i - 4 * FRAG] = f2bf(0.f);
    else if (i < 12 * FRAG)      frag_v[i - 8 * FRAG] = f2bf(0.f);
    else                         frag_att[i - 12 * FRAG] = f2bf(0.f);
  }
  __syncthreads();

  // ---------------- Phase 1: att = Q^T K  (64x64 padded, K = 1024) ----------------
  // 16 tiles of 16x16; wave w owns tiles 2w, 2w+1 (same i-tile -> shared A fragment)
  const int it  = (2 * wv) >> 2;
  const int jt0 = (2 * wv) & 3;
  const int jt1 = (2 * wv + 1) & 3;
  v8f acc0 = {}; v8f acc1 = {};

  for (int ch = 0; ch < CH; ch += CK) {
    // Coalesced reads of the contiguous 1568-float chunk; scatter into frag layout.
    for (int idx = tid; idx < CK * HW; idx += NTHREADS) {
      int cl = idx / HW, ii = idx - cl * HW;
      float qv = q1[base + (size_t)ch * HW + idx];
      float kv = k1[base + (size_t)ch * HW + idx];
      frag_q[(ii >> 4) * FRAG + a_slot(ii & 15, cl)] = f2bf(qv);
      frag_k[(ii >> 4) * FRAG + b_slot(cl, ii & 15)] = f2bf(kv);
    }
    if (ch + CK < CH) {  // gfx1250 global_prefetch_b8 for the next chunk
      __builtin_prefetch(&q1[base + (size_t)(ch + CK) * HW + tid * 6], 0, 0);
      __builtin_prefetch(&k1[base + (size_t)(ch + CK) * HW + tid * 6], 0, 0);
    }
    __syncthreads();

    v16bf a  = *(const v16bf*)&frag_q[it  * FRAG + lane * 16];
    v16bf b0 = *(const v16bf*)&frag_k[jt0 * FRAG + lane * 16];
    v16bf b1 = *(const v16bf*)&frag_k[jt1 * FRAG + lane * 16];
    acc0 = __builtin_amdgcn_wmma_f32_16x16x32_bf16(false, a, false, b0,
                                                   (short)0, acc0, false, false);
    acc1 = __builtin_amdgcn_wmma_f32_16x16x32_bf16(false, a, false, b1,
                                                   (short)0, acc1, false, false);
    __syncthreads();
  }
  {
    // D layout: lane -> N = lane%16 ; VGPR r -> M = r + 8*(lane/16)
    const int mrow = it * 16 + 8 * (lane >> 4);
    const int n0 = jt0 * 16 + (lane & 15);
    const int n1 = jt1 * 16 + (lane & 15);
#pragma unroll
    for (int r = 0; r < 8; ++r) {
      lds_att[(mrow + r) * HWP + n0] = acc0[r];
      lds_att[(mrow + r) * HWP + n1] = acc1[r];
    }
  }
  __syncthreads();

  // ---------------- Phase 2: softmax over the flat 49*49 entries ----------------
  float lmax = -3.4e38f;
  for (int idx = tid; idx < HW * HW; idx += NTHREADS) {
    int ii = idx / HW, jj = idx - ii * HW;
    lmax = fmaxf(lmax, lds_att[ii * HWP + jj]);
  }
#pragma unroll
  for (int off = 16; off; off >>= 1) lmax = fmaxf(lmax, __shfl_xor(lmax, off, 32));
  if (lane == 0) red[wv] = lmax;
  __syncthreads();
  float gmax = red[0];
#pragma unroll
  for (int i = 1; i < 8; ++i) gmax = fmaxf(gmax, red[i]);
  __syncthreads();

  float lsum = 0.f;
  for (int idx = tid; idx < HW * HW; idx += NTHREADS) {
    int ii = idx / HW, jj = idx - ii * HW;
    float e = __expf(lds_att[ii * HWP + jj] - gmax);
    lds_att[ii * HWP + jj] = e;
    lsum += e;
  }
#pragma unroll
  for (int off = 16; off; off >>= 1) lsum += __shfl_xor(lsum, off, 32);
  if (lane == 0) red[wv] = lsum;
  __syncthreads();
  float gsum = 0.f;
#pragma unroll
  for (int i = 0; i < 8; ++i) gsum += red[i];
  const float rinv = 1.0f / gsum;

  // Normalize straight into B-fragment layout (pads already zero).
  for (int idx = tid; idx < HW * HW; idx += NTHREADS) {
    int ii = idx / HW, jj = idx - ii * HW;
    __bf16 w = f2bf(lds_att[ii * HWP + jj] * rinv);
    frag_att[((ii >> 5) * 4 + (jj >> 4)) * FRAG + b_slot(ii & 31, jj & 15)] = w;
  }
  __syncthreads();

  // ---------------- Phase 3: out = V * att + V ----------------
  const int ct = wv >> 2;           // channel sub-tile within 32-ch chunk
  const int jt = wv & 3;            // j-tile
  const unsigned vf_base = (unsigned)(size_t)(void*)lds_vf;

  for (int ch = 0; ch < CH; ch += CK) {
    // Async global -> LDS copy of the f32 V chunk (ASYNCcnt path, no VGPR round trip).
    for (int idx = tid; idx < CK * HW; idx += NTHREADS) {
      int cl = idx / HW, ii = idx - cl * HW;
      unsigned lds_addr = vf_base + (unsigned)(cl * HWP + ii) * 4u;
      unsigned long long ga =
          (unsigned long long)(const void*)&v1[base + (size_t)ch * HW + idx];
      asm volatile("global_load_async_to_lds_b32 %0, %1, off"
                   :: "v"(lds_addr), "v"(ga) : "memory");
    }
    if (ch + CK < CH)
      __builtin_prefetch(&v1[base + (size_t)(ch + CK) * HW + tid * 6], 0, 0);
    asm volatile("s_wait_asynccnt 0x0" ::: "memory");
    __syncthreads();

    // Convert f32 (LDS) -> bf16 A fragments.
    for (int idx = tid; idx < CK * HW; idx += NTHREADS) {
      int cl = idx / HW, ii = idx - cl * HW;
      float vv = lds_vf[cl * HWP + ii];
      frag_v[((cl >> 4) * 2 + (ii >> 5)) * FRAG + a_slot(cl & 15, ii & 31)] = f2bf(vv);
    }
    __syncthreads();

    v8f o = {};
#pragma unroll
    for (int kk = 0; kk < 2; ++kk) {
      v16bf a  = *(const v16bf*)&frag_v[(ct * 2 + kk) * FRAG + lane * 16];
      v16bf bb = *(const v16bf*)&frag_att[(kk * 4 + jt) * FRAG + lane * 16];
      o = __builtin_amdgcn_wmma_f32_16x16x32_bf16(false, a, false, bb,
                                                  (short)0, o, false, false);
    }
    const int crow = ct * 16 + 8 * (lane >> 4);
    const int j = jt * 16 + (lane & 15);
    if (j < HW) {
#pragma unroll
      for (int r = 0; r < 8; ++r) {
        int cc = crow + r;
        out[base + (size_t)(ch + cc) * HW + j] = o[r] + lds_vf[cc * HWP + j];
      }
    }
    __syncthreads();
  }
}

extern "C" void kernel_launch(void* const* d_in, const int* in_sizes, int n_in,
                              void* d_out, int out_size, void* d_ws, size_t ws_size,
                              hipStream_t stream) {
  (void)n_in; (void)out_size; (void)d_ws; (void)ws_size;
  const float* v1 = (const float*)d_in[0];
  const float* q1 = (const float*)d_in[1];
  const float* k1 = (const float*)d_in[2];
  float* out = (float*)d_out;
  const int B = in_sizes[0] / (CH * HW);
  pam_kernel<<<dim3(B), dim3(NTHREADS), 0, stream>>>(v1, q1, k1, out);
}
